// ScoringModule_26070451486929
// MI455X (gfx1250) — matrix operand order repaired and checked by reading, not compiled, not run on hardware
//
#include <hip/hip_runtime.h>
#include <hip/hip_bf16.h>

// ---------------- problem geometry ----------------
constexpr int M_DIM = 4096;   // clusters
constexpr int K_DIM = 3092;   // span size
constexpr int K_PAD = 3104;   // 97 * 32  (bf16 WMMA K-step = 32)
constexpr int N_DIM = 3000;   // hidden
constexpr int N_PAD = 3072;   // 192 * 16
constexpr int FEATN = 20;
constexpr int NBKT  = 10;

// ---------------- workspace layout (bytes) ----------------
constexpr size_t OFF_A    = 0;                                        // bf16 A  [M_DIM][K_PAD]
constexpr size_t OFF_W    = OFF_A   + (size_t)M_DIM * K_PAD * 2;      // bf16 Wt [N_PAD][K_PAD] (K-contiguous)
constexpr size_t OFF_BASE = OFF_W   + (size_t)N_PAD * K_PAD * 2;      // f32 base[N_PAD]
constexpr size_t OFF_WOUT = OFF_BASE + (size_t)N_PAD * 4;             // f32 wout[N_PAD]
constexpr size_t OFF_FEAT = OFF_WOUT + (size_t)N_PAD * 4;             // f32 feat[NBKT][N_PAD]
constexpr size_t OFF_BKT  = OFF_FEAT + (size_t)NBKT * N_PAD * 4;      // i32 bucket[M_DIM]

typedef __attribute__((ext_vector_type(16))) __bf16 v16bf;
typedef __attribute__((ext_vector_type(8)))  float  v8f;

union FragU { uint4 u[2]; v16bf v; };

// ---- CDNA5 async global->LDS path (guarded: falls back to load+ds_store) ----
#if defined(__HIP_DEVICE_COMPILE__)
#if __has_builtin(__builtin_amdgcn_global_load_async_to_lds_b128) && \
    __has_builtin(__builtin_amdgcn_s_wait_asynccnt)
#define USE_ASYNC_LDS 1
#endif
#endif

#ifdef USE_ASYNC_LDS
typedef int v4i_vec __attribute__((vector_size(16)));
typedef __attribute__((address_space(1))) v4i_vec* g_v4i_ptr;
typedef __attribute__((address_space(3))) v4i_vec* l_v4i_ptr;
#endif

__device__ __forceinline__ void copy16_g2l(const void* g, void* l) {
#ifdef USE_ASYNC_LDS
    __builtin_amdgcn_global_load_async_to_lds_b128(
        (g_v4i_ptr)(void*)g, (l_v4i_ptr)l, 0, 0);
#else
    *reinterpret_cast<uint4*>(l) = *reinterpret_cast<const uint4*>(g);
#endif
}

__device__ __forceinline__ void wait_g2l() {
#ifdef USE_ASYNC_LDS
    __builtin_amdgcn_s_wait_asynccnt(0);
#endif
}

// ---------------- prep: A -> bf16, zero-pad K ----------------
__global__ void prep_A(const float* __restrict__ cluster, __bf16* __restrict__ A) {
    size_t i = (size_t)blockIdx.x * blockDim.x + threadIdx.x;
    if (i >= (size_t)M_DIM * K_PAD) return;
    int    k = (int)(i % K_PAD);
    size_t c = i / K_PAD;
    float v = (k < K_DIM) ? cluster[c * K_DIM + k] : 0.0f;
    A[i] = (__bf16)v;
}

// ---------------- prep: Weff = W0[3092+k] + span[k]*W0[6184+k], stored [n][k] bf16 ----------------
__global__ void prep_Wt(const float* __restrict__ W0, const float* __restrict__ span,
                        __bf16* __restrict__ Wt) {
    size_t i = (size_t)blockIdx.x * blockDim.x + threadIdx.x;
    if (i >= (size_t)N_PAD * K_PAD) return;
    int    k = (int)(i % K_PAD);
    size_t n = i / K_PAD;
    float v = 0.0f;
    if (k < K_DIM && n < (size_t)N_DIM) {
        v = W0[(size_t)(3092 + k) * N_DIM + n] + span[k] * W0[(size_t)(6184 + k) * N_DIM + n];
    }
    Wt[i] = (__bf16)v;
}

// ---------------- prep: base[j] = span@W0[0:3092,j] + b0 + genre/seg rows; padded w_out ----------------
__global__ void prep_base(const float* __restrict__ W0, const float* __restrict__ span,
                          const float* __restrict__ b0, const float* __restrict__ genre,
                          const float* __restrict__ seg, const float* __restrict__ wout,
                          float* __restrict__ basep, float* __restrict__ woutp) {
    int j = blockIdx.x * blockDim.x + threadIdx.x;
    if (j >= N_PAD) return;
    float s = 0.0f, wo = 0.0f;
    if (j < N_DIM) {
        s = b0[j];
        for (int k = 0; k < K_DIM; ++k) s += span[k] * W0[(size_t)k * N_DIM + j];
        for (int f = 0; f < FEATN; ++f) {
            s += genre[f] * W0[(size_t)(9296 + f) * N_DIM + j];  // tiled genre rows
            s += seg[f]   * W0[(size_t)(9336 + f) * N_DIM + j];  // segment_distance_emb[0] rows
        }
        wo = wout[j];
    }
    basep[j] = s;
    woutp[j] = wo;
}

// ---------------- prep: feat_tab[b][j] = ad_emb[b] @ W0[9316:9336, j] ----------------
__global__ void prep_feat(const float* __restrict__ W0, const float* __restrict__ ad,
                          float* __restrict__ feat) {
    int j = blockIdx.x * blockDim.x + threadIdx.x;
    int b = blockIdx.y;
    if (j >= N_PAD) return;
    float s = 0.0f;
    if (j < N_DIM)
        for (int f = 0; f < FEATN; ++f) s += ad[b * FEATN + f] * W0[(size_t)(9316 + f) * N_DIM + j];
    feat[(size_t)b * N_PAD + j] = s;
}

// ---------------- prep: distance buckets (exact floor(log2) via clz) + score init ----------------
__global__ void prep_misc(const int* __restrict__ offsets, const float* __restrict__ bout,
                          int* __restrict__ bucket, float* __restrict__ scores) {
    int c = blockIdx.x * blockDim.x + threadIdx.x;
    if (c >= M_DIM) return;
    int d  = offsets[c];
    int ad = d < 0 ? -d : d;
    if (ad < 1) ad = 1;
    int lg = 31 - __clz(ad);
    int bb = (d <= 4) ? d : (lg + 3);
    bb = min(max(bb, 0), 9);
    bucket[c] = bb;
    scores[c] = bout[0];
}

// ---------------- main GEMM ----------------
// Workgroup tile 128(M) x 256(N), 8 waves as 2(M) x 4(N), wave tile 64x64:
// 16 accumulators (128 VGPRs), 16 WMMA per k-step vs 16 ds_load_b128 -> 1:1 ratio,
// each fragment reused 4x. LDS: (8KB A + 16KB B) double-buffered = 48KB.
__global__ __launch_bounds__(256)
void gemm_score(const __bf16* __restrict__ A, const __bf16* __restrict__ Wt,
                const float* __restrict__ basep, const float* __restrict__ woutp,
                const float* __restrict__ feat, const int* __restrict__ bucket,
                float* __restrict__ scores) {
    __shared__ alignas(16) __bf16 Ab[2][128 * 32];   // 8 KB per stage
    __shared__ alignas(16) __bf16 Bb[2][256 * 32];   // 16 KB per stage

    const int tid  = threadIdx.x;
    const int lane = tid & 31;
    const int w    = tid >> 5;        // 8 waves
    const int wm   = w & 1;           // 2 waves along M (64 rows each)
    const int wn   = w >> 1;          // 4 waves along N (64 cols each)
    const int blockM = blockIdx.x * 128;
    const int blockN = blockIdx.y * 256;
    const int KT = K_PAD / 32;        // 97

    auto loadStage = [&](int s, int kt) {
        const int k0 = kt * 32;
        // A tile: 128x32 bf16 = 512 x 16B chunks; 2 per thread
        #pragma unroll
        for (int i = 0; i < 2; ++i) {
            int chunk = tid + i * 256;
            int row = chunk >> 2, kc = chunk & 3;
            copy16_g2l(A + (size_t)(blockM + row) * K_PAD + k0 + kc * 8,
                       &Ab[s][row * 32 + kc * 8]);
        }
        // B tile: 256x32 bf16 = 1024 x 16B chunks; 4 per thread
        #pragma unroll
        for (int i = 0; i < 4; ++i) {
            int chunk = tid + i * 256;
            int row = chunk >> 2, kc = chunk & 3;
            copy16_g2l(Wt + (size_t)(blockN + row) * K_PAD + k0 + kc * 8,
                       &Bb[s][row * 32 + kc * 8]);
        }
    };

    // A fragment (16x32 bf16): lanes 0-15 hold M=l, K {0..7,16..23}; lanes 16-31: K {8..15,24..31}
    auto loadFragA = [&](int s, int mi) -> v16bf {
        int row = wm * 64 + mi * 16 + (lane & 15);
        const uint4* p = reinterpret_cast<const uint4*>(&Ab[s][row * 32 + (lane >> 4) * 8]);
        FragU f; f.u[0] = p[0]; f.u[1] = p[2];
        return f.v;
    };
    // B fragment (32x16 bf16): lanes 0-15 hold N=l, K {0..7,16..23}; lanes 16-31: K {8..15,24..31}
    auto loadFragB = [&](int s, int ni) -> v16bf {
        int row = wn * 64 + ni * 16 + (lane & 15);
        const uint4* p = reinterpret_cast<const uint4*>(&Bb[s][row * 32 + (lane >> 4) * 8]);
        FragU f; f.u[0] = p[0]; f.u[1] = p[2];
        return f.v;
    };

    v8f acc[4][4];
    #pragma unroll
    for (int mi = 0; mi < 4; ++mi)
        #pragma unroll
        for (int ni = 0; ni < 4; ++ni)
            acc[mi][ni] = (v8f){};

    loadStage(0, 0);
    wait_g2l();
    __syncthreads();
    for (int kt = 0; kt < KT; ++kt) {
        const int cur = kt & 1;
        if (kt + 1 < KT) loadStage(cur ^ 1, kt + 1);   // prefetch next stage

        v16bf af[4], bfr[4];
        #pragma unroll
        for (int mi = 0; mi < 4; ++mi) af[mi]  = loadFragA(cur, mi);
        #pragma unroll
        for (int ni = 0; ni < 4; ++ni) bfr[ni] = loadFragB(cur, ni);

        #pragma unroll
        for (int mi = 0; mi < 4; ++mi)
            #pragma unroll
            for (int ni = 0; ni < 4; ++ni)
                acc[mi][ni] = __builtin_amdgcn_wmma_f32_16x16x32_bf16(
                    false, af[mi], false, bfr[ni], (short)0, acc[mi][ni], false, false);

        wait_g2l();
        __syncthreads();
    }

    // Epilogue: D layout = lane n=l%16, VGPR r -> m = r + 8*(l/16)
    const int half = lane >> 4;
    const int nl   = lane & 15;
    #pragma unroll
    for (int mi = 0; mi < 4; ++mi) {
        #pragma unroll
        for (int ni = 0; ni < 4; ++ni) {
            v8f a = acc[mi][ni];
            int ng = blockN + wn * 64 + ni * 16 + nl;
            float bn = basep[ng];
            float wo = woutp[ng];
            #pragma unroll
            for (int r = 0; r < 8; ++r) {
                int mg = blockM + wm * 64 + mi * 16 + r + 8 * half;
                int bk = bucket[mg];
                float v = a[r] + bn + feat[(size_t)bk * N_PAD + ng];
                v = fmaxf(v, 0.0f) * wo;
                // reduce the 16 lanes of this half-wave (one output row)
                v += __shfl_xor(v, 1, 32);
                v += __shfl_xor(v, 2, 32);
                v += __shfl_xor(v, 4, 32);
                v += __shfl_xor(v, 8, 32);
                if (nl == 0) atomicAdd(&scores[mg], v);
            }
        }
    }
}

extern "C" void kernel_launch(void* const* d_in, const int* in_sizes, int n_in,
                              void* d_out, int out_size, void* d_ws, size_t ws_size,
                              hipStream_t stream) {
    const float* span    = (const float*)d_in[0];
    const float* cluster = (const float*)d_in[1];
    const int*   offsets = (const int*)d_in[2];
    const float* genre   = (const float*)d_in[3];
    // d_in[4] same_speaker_emb: unused (speaker_pair is zeroed in reference)
    const float* ad      = (const float*)d_in[5];
    const float* seg     = (const float*)d_in[6];   // row 0 used (segment_distance == 0)
    const float* W0      = (const float*)d_in[7];
    const float* b0      = (const float*)d_in[8];
    const float* wout    = (const float*)d_in[9];
    const float* bout    = (const float*)d_in[10];
    float* scores = (float*)d_out;

    char* ws = (char*)d_ws;
    __bf16* Abf   = (__bf16*)(ws + OFF_A);
    __bf16* Wtbf  = (__bf16*)(ws + OFF_W);
    float*  basep = (float*)(ws + OFF_BASE);
    float*  woutp = (float*)(ws + OFF_WOUT);
    float*  featp = (float*)(ws + OFF_FEAT);
    int*    bkt   = (int*)(ws + OFF_BKT);

    {   // A -> bf16 padded
        size_t n = (size_t)M_DIM * K_PAD;
        prep_A<<<dim3((unsigned)((n + 255) / 256)), dim3(256), 0, stream>>>(cluster, Abf);
    }
    {   // Weff transposed bf16
        size_t n = (size_t)N_PAD * K_PAD;
        prep_Wt<<<dim3((unsigned)((n + 255) / 256)), dim3(256), 0, stream>>>(W0, span, Wtbf);
    }
    prep_base<<<dim3(N_PAD / 256), dim3(256), 0, stream>>>(W0, span, b0, genre, seg, wout,
                                                           basep, woutp);
    prep_feat<<<dim3(N_PAD / 256, NBKT), dim3(256), 0, stream>>>(W0, ad, featp);
    prep_misc<<<dim3(M_DIM / 256), dim3(256), 0, stream>>>(offsets, bout, bkt, scores);

    dim3 grid(M_DIM / 128, N_PAD / 256);   // 32 x 12
    gemm_score<<<grid, dim3(256), 0, stream>>>(Abf, Wtbf, basep, woutp, featp, bkt, scores);
}
